// GraphOU_15264313770622
// MI455X (gfx1250) — matrix operand order repaired and checked by reading, not compiled, not run on hardware
//
#include <hip/hip_runtime.h>

// GraphOU scan on gfx1250:
//  - persistent kernel: 16 WGs x 256 threads, entire T=256 scan in one launch
//  - per-step GEMM agg = adjT @ x via v_wmma_f32_16x16x32_bf16
//  - adjT exact in bf16 (entries are 0/1); x carried as hi+lo bf16 split
//  - 64x1024 bf16 A-tile per WG cached in 129KB LDS via async global->LDS copies
//  - feat/x state in registers (C-fragment element ownership is static)
//  - one device barrier per step (atomic counter + fences + cluster-barrier hint)

#define S_   64
#define T_   256
#define N_   1024
#define TM1  255
#define RPAD 1032          // LDS row pitch in halves (1024 + 8 pad -> conflict-free b128)

typedef __attribute__((ext_vector_type(16))) __bf16 v16bf;
typedef __attribute__((ext_vector_type(8)))  float  v8f;

__device__ __forceinline__ unsigned short f2bf(float f) {
  unsigned u = __float_as_uint(f);
  u += 0x7fffu + ((u >> 16) & 1u);           // round-to-nearest-even
  return (unsigned short)(u >> 16);
}
__device__ __forceinline__ float bf2f(unsigned short h) {
  return __uint_as_float(((unsigned)h) << 16);
}
// pack two floats into (lo16=a, hi16=b) bf16 pair
__device__ __forceinline__ unsigned pack_bf16(float a, float b) {
#if __has_builtin(__builtin_amdgcn_cvt_pk_bf16_f32)
  auto r = __builtin_amdgcn_cvt_pk_bf16_f32(a, b);
  return __builtin_bit_cast(unsigned, r);
#else
  return (unsigned)f2bf(a) | ((unsigned)f2bf(b) << 16);
#endif
}

// ---------- prologue kernels ----------

__global__ void k_prep_adj(const float* __restrict__ adj, unsigned short* __restrict__ adjT) {
  int idx = blockIdx.x * 256 + threadIdx.x;        // N*N threads
  int m = idx >> 10, k = idx & 1023;
  adjT[(size_t)m * N_ + k] = f2bf(adj[(size_t)k * N_ + m]);   // A[m,k] = adj[k,m]
}

__global__ void k_coef(const float* __restrict__ adj, float* __restrict__ coef) {
  int m = blockIdx.x * 256 + threadIdx.x;          // N threads
  float d = 0.f;
  for (int k = 0; k < N_; k++) d += adj[(size_t)k * N_ + m];  // in-degree of node m
  coef[m] = (d > 0.f) ? (0.5f * 0.7f) / (d + 1e-7f) : 0.f;    // OMEGA*GAMMA/(deg+1e-7)
}

__global__ void k_escale(const float* __restrict__ tt, const float* __restrict__ speed,
                         const float* __restrict__ vola,
                         float* __restrict__ eArr, float* __restrict__ sArr) {
  int i = blockIdx.x * 256 + threadIdx.x;
  if (i >= S_ * TM1) return;
  int s = i / TM1, j = i - s * TM1;
  float dt = tt[s * T_ + j + 1] - tt[s * T_ + j];
  float sp = speed[s], vo = vola[s];
  eArr[i] = __expf(-sp * dt);
  float e2 = __expf(-2.f * sp * dt);
  sArr[i] = __fsqrt_rn(vo * vo * (1.f - e2) / (2.f * sp + 1e-6f));
}

__global__ void k_init(const float* __restrict__ x0, float* __restrict__ out,
                       unsigned* __restrict__ Bhi, unsigned* __restrict__ Blo,
                       unsigned* __restrict__ counter) {
  int i = blockIdx.x * 256 + threadIdx.x;          // 65536 threads
  if (i < S_ * N_) out[(size_t)(i >> 10) * (T_ * N_) + (i & 1023)] = x0[i & 1023];
  if (i < 512 * 64) {                               // pack x0 into B-buffer 0 (K-pair dwords)
    int p = i >> 6;
    float a = x0[2 * p], b = x0[2 * p + 1];
    unsigned ph = pack_bf16(a, b);
    float ahf = __uint_as_float(ph << 16);
    float bhf = __uint_as_float(ph & 0xffff0000u);
    Bhi[i] = ph;
    Blo[i] = pack_bf16(a - ahf, b - bhf);
  }
  if (i == 0) *counter = 0u;
}

// ---------- persistent simulation kernel ----------

__global__ __launch_bounds__(256)
void k_sim(const unsigned short* __restrict__ adjT,
           const float* __restrict__ coef,
           const float* __restrict__ eArr,
           const float* __restrict__ sArr,
           const float* __restrict__ noise,
           const int*   __restrict__ items,
           const float* __restrict__ x0,
           unsigned* __restrict__ Bhi,
           unsigned* __restrict__ Blo,
           unsigned* __restrict__ counter,
           float* __restrict__ out) {
  __shared__ unsigned short Alds[64 * RPAD];       // 129 KB: this WG's adjT slab, bf16

  const int wg     = blockIdx.x;                   // 0..15 -> rows [wg*64, wg*64+64)
  const int tid    = threadIdx.x;
  const int wave   = tid >> 5;
  const int lane   = tid & 31;
  const int laneHi = lane >> 4;
  const int lane16 = lane & 15;

  // Stage this WG's 64x1024 bf16 A tile into LDS with async global->LDS B128
  // copies (ASYNCcnt path; 32 outstanding per thread, max counter is 63).
  {
    const char* gbase = (const char*)(adjT + (size_t)wg * 64 * N_);
    for (int i = tid; i < 64 * (N_ / 8); i += 256) {
      int row = i >> 7, c = i & 127;               // 128 x 16B chunks per row
      unsigned lds_off = (unsigned)(uintptr_t)(&Alds[row * RPAD + c * 8]);
      unsigned long long gaddr =
          (unsigned long long)(uintptr_t)gbase + (unsigned long long)(row * 128 + c) * 16ull;
      asm volatile("global_load_async_to_lds_b128 %0, %1, off"
                   :: "v"(lds_off), "v"(gaddr) : "memory");
    }
    asm volatile("s_wait_asynccnt 0x0" ::: "memory");
  }
  __syncthreads();

  // Wave -> (M-subtile, two S-subtiles). Lane -> fixed (m,s) elements per C layout.
  const int mt    = wave >> 1;                     // 0..3
  const int m0    = wg * 64 + mt * 16;
  const int mBase = m0 + laneHi * 8;               // this lane's 8 consecutive rows
  const int sA    = (wave & 1) * 32 + lane16;      // scenario of tile A column
  const int sB    = sA + 16;                       // scenario of tile B column

  float coefR[8], xA[8], xB[8], attA[8], winA[8], attB[8], winB[8];
  {
    int it0A = items[sA * T_];
    int it0B = items[sB * T_];
#pragma unroll
    for (int j = 0; j < 8; j++) {
      coefR[j] = coef[mBase + j];
      float v = x0[mBase + j];
      xA[j] = v; xB[j] = v;
      attA[j] = (it0A == mBase + j) ? 1.f : 0.f;  winA[j] = 0.f;
      attB[j] = (it0B == mBase + j) ? 1.f : 0.f;  winB[j] = 0.f;
    }
  }

  const unsigned short* Arow = &Alds[(mt * 16 + lane16) * RPAD + laneHi * 16];
  const int pb = laneHi * 8;                       // K-pair offset for B fragment
  const int pw = mBase >> 1;                       // K-pair index of this lane's rows

  for (int t = 1; t < T_; t++) {
    const unsigned* BH = Bhi + ((t - 1) & 1) * (512 * 64);
    const unsigned* BL = Blo + ((t - 1) & 1) * (512 * 64);
    v8f accA = {}; v8f accB = {};

#pragma unroll 2
    for (int k0 = 0; k0 < N_; k0 += 32) {
      union { uint4 q[2]; v16bf v; } af;           // A 16x32 bf16 fragment (contiguous K: element s = k0 + 16*laneHi + s)
      af.q[0] = *(const uint4*)(Arow + k0);        // lanes<16: K k0+0..7 ; lanes>=16: K k0+16..23
      af.q[1] = *(const uint4*)(Arow + k0 + 8);    // lanes<16: K k0+8..15; lanes>=16: K k0+24..31
      const int p0 = (k0 >> 1) + pb;
      union { unsigned u[8]; v16bf v; } bhA, blA, bhB, blB;
#pragma unroll
      for (int v = 0; v < 8; v++) {                // B 32x16: VGPR v = K-pair (2v,2v+1)
        const unsigned* rh = BH + (p0 + v) * 64;
        const unsigned* rl = BL + (p0 + v) * 64;
        bhA.u[v] = rh[sA];  bhB.u[v] = rh[sB];
        blA.u[v] = rl[sA];  blB.u[v] = rl[sB];
      }
      accA = __builtin_amdgcn_wmma_f32_16x16x32_bf16(false, af.v, false, bhA.v, (short)0, accA, false, false);
      accB = __builtin_amdgcn_wmma_f32_16x16x32_bf16(false, af.v, false, bhB.v, (short)0, accB, false, false);
      accA = __builtin_amdgcn_wmma_f32_16x16x32_bf16(false, af.v, false, blA.v, (short)0, accA, false, false);
      accB = __builtin_amdgcn_wmma_f32_16x16x32_bf16(false, af.v, false, blB.v, (short)0, accB, false, false);
    }

    unsigned* WH = Bhi + (t & 1) * (512 * 64);
    unsigned* WL = Blo + (t & 1) * (512 * 64);

#define EPILOGUE(acc, sX, xS, attS, winS)                                        \
    {                                                                            \
      float e  = eArr[(sX) * TM1 + (t - 1)];                                     \
      float sc = sArr[(sX) * TM1 + (t - 1)];                                     \
      int   it = items[(sX) * T_ + t];                                           \
      const float* nz = noise + ((size_t)(sX) * TM1 + (t - 1)) * N_ + mBase;     \
      float4 n0 = *(const float4*)nz;                                            \
      float4 n1 = *(const float4*)(nz + 4);                                      \
      float nv[8] = {n0.x, n0.y, n0.z, n0.w, n1.x, n1.y, n1.z, n1.w};            \
      float o[8];                                                                \
      _Pragma("unroll")                                                          \
      for (int j = 0; j < 8; j++) {                                              \
        float agg   = acc[j];                                                    \
        float ratio = winS[j] * __builtin_amdgcn_rcpf(attS[j] + 1e-6f);          \
        float level = coefR[j] * agg + 0.5f * ratio * ratio;                     \
        float xn    = xS[j] * e + (1.f - e) * level + nv[j] * sc;                \
        xS[j] = xn; o[j] = xn;                                                   \
        if (it == mBase + j) { attS[j] += 1.f; winS[j] += (xn >= 0.f) ? 1.f : 0.f; } \
      }                                                                          \
      float* op = out + (size_t)(sX) * (T_ * N_) + (size_t)t * N_ + mBase;       \
      *(float4*)op       = make_float4(o[0], o[1], o[2], o[3]);                  \
      *(float4*)(op + 4) = make_float4(o[4], o[5], o[6], o[7]);                  \
      _Pragma("unroll")                                                          \
      for (int q = 0; q < 4; q++) {                                              \
        float a = o[2 * q], b = o[2 * q + 1];                                    \
        unsigned ph = pack_bf16(a, b);                                           \
        float ahf = __uint_as_float(ph << 16);                                   \
        float bhf = __uint_as_float(ph & 0xffff0000u);                           \
        WH[(pw + q) * 64 + (sX)] = ph;                                           \
        WL[(pw + q) * 64 + (sX)] = pack_bf16(a - ahf, b - bhf);                  \
      }                                                                          \
    }

    EPILOGUE(accA, sA, xA, attA, winA)
    EPILOGUE(accB, sB, xB, attB, winB)
#undef EPILOGUE

    // Device-wide barrier across the 16 WGs (monotonic counter, release/acquire).
    __threadfence();
    __syncthreads();
    if (tid == 0) {
      atomicAdd(counter, 1u);
      const unsigned target = 16u * (unsigned)t;
      while (atomicAdd(counter, 0u) < target) __builtin_amdgcn_s_sleep(1);
    }
    __syncthreads();
    __threadfence();
    // Cluster-barrier hint: NOP when not cluster-dispatched; exact sync if the
    // 16-WG grid is launched as one workgroup-cluster on CDNA5.
    __builtin_amdgcn_s_cluster_barrier();
  }
}

// ---------- host side ----------

extern "C" void kernel_launch(void* const* d_in, const int* in_sizes, int n_in,
                              void* d_out, int out_size, void* d_ws, size_t ws_size,
                              hipStream_t stream) {
  (void)in_sizes; (void)n_in; (void)out_size; (void)ws_size;
  const float* x0    = (const float*)d_in[0];
  const float* tArr  = (const float*)d_in[1];
  const float* speed = (const float*)d_in[2];
  const float* vola  = (const float*)d_in[3];
  const float* adj   = (const float*)d_in[4];
  const float* noise = (const float*)d_in[5];
  const int*   items = (const int*)d_in[6];
  float* out = (float*)d_out;

  char* ws = (char*)d_ws;
  unsigned short* adjT  = (unsigned short*)(ws + 0);        // 2,097,152 B
  float*    coef    = (float*)(ws + 2097152);               //     4,096 B
  float*    eArr    = (float*)(ws + 2101248);               //    65,536 B (uses 65,280)
  float*    sArr    = (float*)(ws + 2166784);               //    65,536 B
  unsigned* Bhi     = (unsigned*)(ws + 2232320);            //   262,144 B (2 buffers)
  unsigned* Blo     = (unsigned*)(ws + 2494464);            //   262,144 B
  unsigned* counter = (unsigned*)(ws + 2756608);            //         4 B

  k_prep_adj<<<(N_ * N_) / 256, 256, 0, stream>>>(adj, adjT);
  k_coef<<<N_ / 256, 256, 0, stream>>>(adj, coef);
  k_escale<<<(S_ * TM1 + 255) / 256, 256, 0, stream>>>(tArr, speed, vola, eArr, sArr);
  k_init<<<(S_ * N_) / 256, 256, 0, stream>>>(x0, out, Bhi, Blo, counter);
  k_sim<<<16, 256, 0, stream>>>(adjT, coef, eArr, sArr, noise, items, x0,
                                Bhi, Blo, counter, out);
}
